// MaskedAttention_89833535963459
// MI455X (gfx1250) — matrix operand order repaired and checked by reading, not compiled, not run on hardware
//
#include <hip/hip_runtime.h>
#include <float.h>

#define DIM 768
#define NHEAD 12
#define HDIM 64
#define BATCH 4
#define SEQ 1024
#define TOKENS (BATCH * SEQ)   // 4096
#define QKVO (3 * DIM)         // 2304
#define ATT_SCALE 0.125f       // 64^-0.5

typedef __bf16 bf16_t;
typedef __attribute__((ext_vector_type(16))) __bf16 v16bf;
typedef __attribute__((ext_vector_type(8)))  float  v8f;
typedef __attribute__((ext_vector_type(4)))  float  f32x4;
typedef __attribute__((ext_vector_type(2)))  unsigned int u32x2;

union FragU {
  v16bf v;
  struct { f32x4 lo, hi; } f;
};

#define WMMA_BF16(A, B, C) \
  __builtin_amdgcn_wmma_f32_16x16x32_bf16(false, (A), false, (B), (short)0, (C), false, false)

// ---------------------------------------------------------------------------
// Fragment loaders matching the CDNA5 WMMA VGPR layouts (wave32).
// A (16x32 bf16, MxK), source row-major [M][K] with row stride ld:
//   lane l: m = l&15, kh = l>>4; elems j<8 -> k=kh*8+j ; j>=8 -> k=16+kh*8+(j-8)
__device__ __forceinline__ v16bf load_frag_a(const bf16_t* __restrict__ base, int ld) {
  const int lane = threadIdx.x & 31;
  const int m = lane & 15;
  const int kh = lane >> 4;
  const bf16_t* p = base + (size_t)m * ld + kh * 8;
  FragU u;
  u.f.lo = *reinterpret_cast<const f32x4*>(p);
  u.f.hi = *reinterpret_cast<const f32x4*>(p + 16);
  return u.v;
}

// B (32x16 bf16, KxN) where memory holds Bsrc[n][k] row-major (k contiguous):
//   lane l: n = l&15, kh = l>>4; elems j<8 -> k=kh*8+j ; j>=8 -> k=16+kh*8+(j-8)
__device__ __forceinline__ v16bf load_frag_b(const bf16_t* __restrict__ base, int ld) {
  const int lane = threadIdx.x & 31;
  const int n = lane & 15;
  const int kh = lane >> 4;
  const bf16_t* p = base + (size_t)n * ld + kh * 8;
  FragU u;
  u.f.lo = *reinterpret_cast<const f32x4*>(p);
  u.f.hi = *reinterpret_cast<const f32x4*>(p + 16);
  return u.v;
}

// ---------------------------------------------------------------------------
// fp32 -> bf16 conversion, 4 elements per thread
__global__ __launch_bounds__(256) void cvt_bf16_kernel(const float* __restrict__ src,
                                                       bf16_t* __restrict__ dst, int n4) {
  int i = blockIdx.x * blockDim.x + threadIdx.x;
  if (i >= n4) return;
  f32x4 v = reinterpret_cast<const f32x4*>(src)[i];
  union { bf16_t h[4]; u32x2 u; } o;
  o.h[0] = (bf16_t)v.x; o.h[1] = (bf16_t)v.y;
  o.h[2] = (bf16_t)v.z; o.h[3] = (bf16_t)v.w;
  *reinterpret_cast<u32x2*>(dst + (size_t)i * 4) = o.u;
}

// ---------------------------------------------------------------------------
// QKV GEMM, register-blocked 32x64 per wave:
// [4096,768]bf16 x [2304,768]^T bf16 -> Q,K [B,H,N,D], V^T [B,H,D,N]
// An N-group of 64 channels is exactly one head's D-slice of q, k, or v.
__global__ __launch_bounds__(256) void qkv_gemm(const bf16_t* __restrict__ xb,
                                                const bf16_t* __restrict__ wb,
                                                bf16_t* __restrict__ qbuf,
                                                bf16_t* __restrict__ kbuf,
                                                bf16_t* __restrict__ vtbuf) {
  const int wave = (blockIdx.x * blockDim.x + threadIdx.x) >> 5;
  const int NG = QKVO / 64;                  // 36 N-groups
  const int mt = wave / NG;                  // 0..127 (32-row tiles)
  const int ng = wave - mt * NG;
  const int lane = threadIdx.x & 31;
  const int n = lane & 15;
  const int half = lane >> 4;

  v8f acc[2][4] = {};
  const bf16_t* arow0 = xb + (size_t)(mt * 32) * DIM;
  const bf16_t* arow1 = arow0 + (size_t)16 * DIM;
  const bf16_t* brow = wb + (size_t)(ng * 64) * DIM;

  for (int k = 0; k < DIM; k += 32) {
    v16bf a0 = load_frag_a(arow0 + k, DIM);
    v16bf a1 = load_frag_a(arow1 + k, DIM);
#pragma unroll
    for (int j = 0; j < 4; ++j) {
      v16bf b = load_frag_b(brow + (size_t)(j * 16) * DIM + k, DIM);
      acc[0][j] = WMMA_BF16(a0, b, acc[0][j]);
      acc[1][j] = WMMA_BF16(a1, b, acc[1][j]);
    }
  }

  const int obase = ng * 64;                 // output channel base
  const int sel = obase / DIM;               // 0=q 1=k 2=v
  const int oo = obase - sel * DIM;          // multiple of 64
  const int h = oo / HDIM;                   // head index
#pragma unroll
  for (int mi = 0; mi < 2; ++mi) {
#pragma unroll
    for (int r = 0; r < 8; ++r) {
      const int mrow = mt * 32 + mi * 16 + r + half * 8;
      const int bidx = mrow >> 10;
      const int tok = mrow & (SEQ - 1);
      const size_t bh = (size_t)bidx * NHEAD + h;
#pragma unroll
      for (int j = 0; j < 4; ++j) {
        const int d = j * 16 + n;
        const bf16_t val = (bf16_t)acc[mi][j][r];
        if (sel == 0)      qbuf[(bh * SEQ + tok) * HDIM + d] = val;
        else if (sel == 1) kbuf[(bh * SEQ + tok) * HDIM + d] = val;
        else               vtbuf[(bh * HDIM + d) * SEQ + tok] = val;
      }
    }
  }
}

// ---------------------------------------------------------------------------
// Flash-style masked attention: one wave per (b,h,16-query tile)
__global__ __launch_bounds__(256) void attn_kernel(const bf16_t* __restrict__ qb,
                                                   const bf16_t* __restrict__ kb,
                                                   const bf16_t* __restrict__ vtb,
                                                   const unsigned char* __restrict__ mask,
                                                   bf16_t* __restrict__ hb) {
  __shared__ float plds[8][16][33];          // per-wave P staging, padded stride
  const int wflat = (blockIdx.x * blockDim.x + threadIdx.x) >> 5;
  const int wl = threadIdx.x >> 5;
  const int lane = threadIdx.x & 31;
  const int n = lane & 15;
  const int half = lane >> 4;

  const int qt = wflat & 63;                 // query tile within head
  const int bh = wflat >> 6;                 // 0..47
  const int b = bh / NHEAD;
  const int h = bh - b * NHEAD;

  const bf16_t* qbase = qb + ((size_t)bh * SEQ + qt * 16) * HDIM;
  const v16bf qa0 = load_frag_a(qbase, HDIM);       // d = 0..31
  const v16bf qa1 = load_frag_a(qbase + 32, HDIM);  // d = 32..63

  const unsigned char* bm = mask + (size_t)b * SEQ;
  bool qm[8];
#pragma unroll
  for (int r = 0; r < 8; ++r) qm[r] = bm[qt * 16 + r + half * 8] != 0;

  float mrun[8], lrun[8];
#pragma unroll
  for (int r = 0; r < 8; ++r) { mrun[r] = -FLT_MAX; lrun[r] = 0.0f; }
  v8f o0 = {}, o1 = {}, o2 = {}, o3 = {};

  const bf16_t* krow = kb + (size_t)bh * SEQ * HDIM;
  const bf16_t* vrow = vtb + (size_t)bh * HDIM * SEQ;
  float (*pl)[33] = plds[wl];

  for (int k0 = 0; k0 < SEQ; k0 += 32) {
    if (k0 + 32 < SEQ) {
      __builtin_prefetch(krow + (size_t)(k0 + 32) * HDIM, 0, 0);
      __builtin_prefetch(vrow + (k0 + 32), 0, 0);
    }
    // S = Q K^T  (two 16-key sub-tiles, K-dim = 64 -> 2 frags each)
    const v16bf kf00 = load_frag_b(krow + (size_t)k0 * HDIM, HDIM);
    const v16bf kf01 = load_frag_b(krow + (size_t)k0 * HDIM + 32, HDIM);
    const v16bf kf10 = load_frag_b(krow + (size_t)(k0 + 16) * HDIM, HDIM);
    const v16bf kf11 = load_frag_b(krow + (size_t)(k0 + 16) * HDIM + 32, HDIM);
    v8f s0 = {}, s1 = {};
    s0 = WMMA_BF16(qa0, kf00, s0);
    s0 = WMMA_BF16(qa1, kf01, s0);
    s1 = WMMA_BF16(qa0, kf10, s1);
    s1 = WMMA_BF16(qa1, kf11, s1);

    const bool km0 = bm[k0 + n] != 0;
    const bool km1 = bm[k0 + 16 + n] != 0;

#pragma unroll
    for (int r = 0; r < 8; ++r) {
      float a0 = (km0 && qm[r]) ? s0[r] * ATT_SCALE : -FLT_MAX;
      float a1 = (km1 && qm[r]) ? s1[r] * ATT_SCALE : -FLT_MAX;
      float mx = fmaxf(a0, a1);
#pragma unroll
      for (int off = 1; off < 16; off <<= 1) mx = fmaxf(mx, __shfl_xor(mx, off, 32));
      const float mnew = fmaxf(mrun[r], mx);
      const float alpha = __expf(mrun[r] - mnew);
      const float p0 = __expf(a0 - mnew);
      const float p1 = __expf(a1 - mnew);
      float ps = p0 + p1;
#pragma unroll
      for (int off = 1; off < 16; off <<= 1) ps += __shfl_xor(ps, off, 32);
      lrun[r] = lrun[r] * alpha + ps;
      mrun[r] = mnew;
      o0[r] *= alpha; o1[r] *= alpha; o2[r] *= alpha; o3[r] *= alpha;
      pl[r + half * 8][n] = p0;            // C-layout -> LDS
      pl[r + half * 8][16 + n] = p1;
    }
    asm volatile("s_wait_dscnt 0x0" ::: "memory");  // wave-private LDS RAW
    // read back in A-layout, convert to bf16
    v16bf pa;
#pragma unroll
    for (int j = 0; j < 8; ++j)  pa[j] = (bf16_t)pl[n][half * 8 + j];
#pragma unroll
    for (int j = 8; j < 16; ++j) pa[j] = (bf16_t)pl[n][16 + half * 8 + (j - 8)];

    // O += P V  (V^T rows d, token-contiguous -> perfect B frags)
    o0 = WMMA_BF16(pa, load_frag_b(vrow + (size_t)0 * 16 * SEQ + k0, SEQ), o0);
    o1 = WMMA_BF16(pa, load_frag_b(vrow + (size_t)1 * 16 * SEQ + k0, SEQ), o1);
    o2 = WMMA_BF16(pa, load_frag_b(vrow + (size_t)2 * 16 * SEQ + k0, SEQ), o2);
    o3 = WMMA_BF16(pa, load_frag_b(vrow + (size_t)3 * 16 * SEQ + k0, SEQ), o3);
  }

  bf16_t* hrow = hb + ((size_t)b * SEQ + qt * 16) * DIM + h * HDIM;
#pragma unroll
  for (int r = 0; r < 8; ++r) {
    const float inv = 1.0f / lrun[r];
    bf16_t* dst = hrow + (size_t)(r + half * 8) * DIM + n;
    dst[0]  = (bf16_t)(o0[r] * inv);
    dst[16] = (bf16_t)(o1[r] * inv);
    dst[32] = (bf16_t)(o2[r] * inv);
    dst[48] = (bf16_t)(o3[r] * inv);
  }
}

// ---------------------------------------------------------------------------
// Output projection, register-blocked 32x64 per wave:
// [4096,768]bf16 x [768,768]^T + bias -> f32 out
__global__ __launch_bounds__(256) void proj_gemm(const bf16_t* __restrict__ hb,
                                                 const bf16_t* __restrict__ wb,
                                                 const float* __restrict__ bias,
                                                 float* __restrict__ out) {
  const int wave = (blockIdx.x * blockDim.x + threadIdx.x) >> 5;
  const int NG = DIM / 64;                   // 12 N-groups
  const int mt = wave / NG;                  // 0..127
  const int ng = wave - mt * NG;
  const int lane = threadIdx.x & 31;
  const int n = lane & 15;
  const int half = lane >> 4;

  v8f acc[2][4] = {};
  const bf16_t* arow0 = hb + (size_t)(mt * 32) * DIM;
  const bf16_t* arow1 = arow0 + (size_t)16 * DIM;
  const bf16_t* brow = wb + (size_t)(ng * 64) * DIM;

  for (int k = 0; k < DIM; k += 32) {
    v16bf a0 = load_frag_a(arow0 + k, DIM);
    v16bf a1 = load_frag_a(arow1 + k, DIM);
#pragma unroll
    for (int j = 0; j < 4; ++j) {
      v16bf b = load_frag_b(brow + (size_t)(j * 16) * DIM + k, DIM);
      acc[0][j] = WMMA_BF16(a0, b, acc[0][j]);
      acc[1][j] = WMMA_BF16(a1, b, acc[1][j]);
    }
  }

#pragma unroll
  for (int mi = 0; mi < 2; ++mi) {
#pragma unroll
    for (int j = 0; j < 4; ++j) {
      const int o = ng * 64 + j * 16 + n;
      const float bv = bias[o];
#pragma unroll
      for (int r = 0; r < 8; ++r) {
        out[(size_t)(mt * 32 + mi * 16 + r + half * 8) * DIM + o] = acc[mi][j][r] + bv;
      }
    }
  }
}

// ---------------------------------------------------------------------------
extern "C" void kernel_launch(void* const* d_in, const int* in_sizes, int n_in,
                              void* d_out, int out_size, void* d_ws, size_t ws_size,
                              hipStream_t stream) {
  (void)in_sizes; (void)n_in; (void)out_size; (void)ws_size;
  const float* x          = (const float*)d_in[0];
  const unsigned char* mk = (const unsigned char*)d_in[1];  // bool, 1 byte
  const float* w_qkv      = (const float*)d_in[2];
  const float* w_proj     = (const float*)d_in[3];
  const float* b_proj     = (const float*)d_in[4];
  float* out              = (float*)d_out;

  char* ws = (char*)d_ws;
  size_t off = 0;
  auto carve = [&](size_t elems) -> bf16_t* {
    bf16_t* p = (bf16_t*)(ws + off);
    off += (elems * sizeof(bf16_t) + 255) & ~(size_t)255;
    return p;
  };
  bf16_t* xb     = carve((size_t)TOKENS * DIM);
  bf16_t* wqkvb  = carve((size_t)QKVO * DIM);
  bf16_t* wprojb = carve((size_t)DIM * DIM);
  bf16_t* qbuf   = carve((size_t)TOKENS * DIM);  // [B,H,N,D]
  bf16_t* kbuf   = carve((size_t)TOKENS * DIM);  // [B,H,N,D]
  bf16_t* vtbuf  = carve((size_t)TOKENS * DIM);  // [B,H,D,N]
  bf16_t* hbuf   = carve((size_t)TOKENS * DIM);  // [B*N, C]

  auto cvt = [&](const float* s, bf16_t* d, int elems) {
    int n4 = elems / 4;
    cvt_bf16_kernel<<<(n4 + 255) / 256, 256, 0, stream>>>(s, d, n4);
  };
  cvt(x, xb, TOKENS * DIM);
  cvt(w_qkv, wqkvb, QKVO * DIM);
  cvt(w_proj, wprojb, DIM * DIM);

  {
    int waves = (TOKENS / 32) * (QKVO / 64);   // 4608
    qkv_gemm<<<waves / 8, 256, 0, stream>>>(xb, wqkvb, qbuf, kbuf, vtbuf);
  }
  {
    int waves = BATCH * NHEAD * (SEQ / 16);    // 3072
    attn_kernel<<<waves / 8, 256, 0, stream>>>(qbuf, kbuf, vtbuf, mk, hbuf);
  }
  {
    int waves = (TOKENS / 32) * (DIM / 64);    // 1536
    proj_gemm<<<waves / 8, 256, 0, stream>>>(hbuf, wprojb, b_proj, out);
  }
}
